// Encoder_41412074668228
// MI455X (gfx1250) — matrix operand order repaired and checked by reading, not compiled, not run on hardware
//
#include <hip/hip_runtime.h>
#include <stdint.h>

typedef __attribute__((ext_vector_type(16))) __bf16 v16bf;
typedef __attribute__((ext_vector_type(8)))  float  v8f;
typedef __attribute__((ext_vector_type(4)))  float  fvec4;   // builtin-compatible float4

#define FEAT 128
#define MT   64    // rows per block
#define LD   136   // bf16 element stride for LDS tiles (128 + 8 pad -> de-conflicts banks)

__device__ __forceinline__ uint32_t f2bf(float f) {
  // round-to-nearest-even fp32 -> bf16, result in low 16 bits
  uint32_t u = __float_as_uint(f);
  uint32_t r = u + 0x7FFFu + ((u >> 16) & 1u);
  return r >> 16;
}

union FragU {
  uint4 q[2];
  v16bf v;
};

// A/B fragment = 8 VGPRs = 16 bf16 per lane; per-lane data is two 16 B chunks
// at k0..k0+7 and k0+16..k0+23 (per the CDNA5 16-bit A-matrix VGPR table).
__device__ __forceinline__ v16bf load_frag_lds(const unsigned short* sm, int elemOff) {
  FragU f;
  f.q[0] = *(const uint4*)(sm + elemOff);
  f.q[1] = *(const uint4*)(sm + elemOff + 16);
  return f.v;
}

__device__ __forceinline__ v16bf load_frag_ws(const uint32_t* wsFrag, int lane) {
  FragU f;
  const uint4* p = (const uint4*)wsFrag + lane * 2;   // 32 B per lane, contiguous
  f.q[0] = p[0];
  f.q[1] = p[1];
  return f.v;
}

// channel weights folded into W1 columns: self(0..127)=1, n0*1.0, n1*0.5, n2*2.0
__device__ __forceinline__ float chscale(int k) {
  return (k < 256) ? 1.0f : (k < 384) ? 0.5f : 2.0f;
}

// Pre-swizzle W1 (fp32 [128,512] row-major, torch Linear weight) into bf16
// B-matrix fragments for V_WMMA_F32_16X16X32_BF16: B[k][n] = W1[n][k]*chscale(k).
// Fragment (ks, nt): K rows ks*32..+31, N cols nt*16..+15.
// Lane L: n = nt*16 + L%16, khalf = L/16; dword v packs K = khalf*8 + (v/4)*16 + 2*(v%4), +1.
// Storage: ws[frag*256 + lane*8 + v]  (32 B contiguous per lane).
__global__ void prep_w1(const float* __restrict__ W1, uint32_t* __restrict__ ws) {
  int t = blockIdx.x * blockDim.x + threadIdx.x;
  if (t >= 128 * 32) return;               // 16 ksteps * 8 ntiles = 128 fragments
  int frag = t >> 5, lane = t & 31;
  int ks = frag >> 3, nt = frag & 7;
  int n = nt * 16 + (lane & 15);
  int khalf = lane >> 4;
  uint32_t* dst = ws + frag * 256 + lane * 8;
#pragma unroll
  for (int v = 0; v < 8; ++v) {
    int k = ks * 32 + khalf * 8 + (v >> 2) * 16 + 2 * (v & 3);
    float f0 = W1[n * 512 + k]     * chscale(k);
    float f1 = W1[n * 512 + k + 1] * chscale(k + 1);
    dst[v] = f2bf(f0) | (f2bf(f1) << 16);
  }
}

// Same for W2 (fp32 [128,128]), no scaling. 4 ksteps * 8 ntiles = 32 fragments.
__global__ void prep_w2(const float* __restrict__ W2, uint32_t* __restrict__ ws) {
  int t = blockIdx.x * blockDim.x + threadIdx.x;
  if (t >= 32 * 32) return;
  int frag = t >> 5, lane = t & 31;
  int ks = frag >> 3, nt = frag & 7;
  int n = nt * 16 + (lane & 15);
  int khalf = lane >> 4;
  uint32_t* dst = ws + frag * 256 + lane * 8;
#pragma unroll
  for (int v = 0; v < 8; ++v) {
    int k = ks * 32 + khalf * 8 + (v >> 2) * 16 + 2 * (v & 3);
    dst[v] = f2bf(W2[n * 128 + k]) | (f2bf(W2[n * 128 + k + 1]) << 16);
  }
}

__global__ __launch_bounds__(256) void encoder_main(
    const int*   __restrict__ nodes,
    const float* __restrict__ features,
    const float* __restrict__ n0,
    const float* __restrict__ n1,
    const float* __restrict__ n2,
    const float* __restrict__ b1,
    const float* __restrict__ b2,
    const uint32_t* __restrict__ wsW1,
    const uint32_t* __restrict__ wsW2,
    float* __restrict__ out,
    int Btot)
{
  // double-buffered input channel tiles + hidden tile (bf16)
  __shared__ __align__(16) unsigned short smA[2][MT * LD];
  __shared__ __align__(16) unsigned short smH[MT * LD];

  const int tid   = threadIdx.x;
  const int lane  = tid & 31;
  const int wav   = tid >> 5;     // 0..7
  const int mrow  = lane & 15;    // A: M row within tile / B,C,D: N column within tile
  const int khalf = lane >> 4;    // K-half selector per WMMA VGPR layout
  const int nt    = wav;          // wave owns one N tile, all four M tiles

  const int rowbase = blockIdx.x * MT;

  // stage channel 0 (gather channel, regular temporal: features can hit L2) into buf 0
#pragma unroll
  for (int it = 0; it < 8; ++it) {
    int linear = (it * 256 + tid) * 4;
    int row = linear >> 7;
    int col = linear & 127;
    int gRow = rowbase + row;
    if (gRow >= Btot) gRow = Btot - 1;
    fvec4 f = *(const fvec4*)(features + (long long)nodes[gRow] * FEAT + col);
    uint2 pk;
    pk.x = f2bf(f.x) | (f2bf(f.y) << 16);
    pk.y = f2bf(f.z) | (f2bf(f.w) << 16);
    *(uint2*)&smA[0][row * LD + col] = pk;
  }
  __syncthreads();

  v8f acc[4];
#pragma unroll
  for (int t = 0; t < 4; ++t) acc[t] = (v8f)0.0f;

  // ---- phase 1: h = tanh(combined @ W1^T + b1), channel-by-channel over K ----
  for (int c = 0; c < 4; ++c) {
    // stage next channel (streamed once -> non-temporal) into the other buffer;
    // independent of this iteration's compute -> scheduler overlaps them.
    if (c < 3) {
      int cc = c + 1;
      const float* nb = (cc == 1) ? n0 : (cc == 2) ? n1 : n2;
      unsigned short* dstbuf = smA[cc & 1];
#pragma unroll
      for (int it = 0; it < 8; ++it) {
        int linear = (it * 256 + tid) * 4;
        int row = linear >> 7;
        int col = linear & 127;
        int gRow = rowbase + row;
        if (gRow >= Btot) gRow = Btot - 1;
        fvec4 f = __builtin_nontemporal_load(
            (const fvec4*)(nb + (long long)gRow * FEAT + col));
        uint2 pk;
        pk.x = f2bf(f.x) | (f2bf(f.y) << 16);
        pk.y = f2bf(f.z) | (f2bf(f.w) << 16);
        *(uint2*)&dstbuf[row * LD + col] = pk;
      }
    }

    const unsigned short* cur = smA[c & 1];
#pragma unroll
    for (int kl = 0; kl < 4; ++kl) {       // 4 x K=32 per 128-wide channel
      // one B fragment per k-step per wave, reused by 4 WMMAs (4 M tiles)
      v16bf b = load_frag_ws(wsW1 + ((c * 4 + kl) * 8 + nt) * 256, lane);
      // preload all 4 A fragments -> 8 ds_load_b128 in flight, partial waits
      v16bf afr[4];
#pragma unroll
      for (int t = 0; t < 4; ++t)
        afr[t] = load_frag_lds(cur, (t * 16 + mrow) * LD + kl * 32 + khalf * 8);
#pragma unroll
      for (int t = 0; t < 4; ++t)
        acc[t] = __builtin_amdgcn_wmma_f32_16x16x32_bf16(
            false, afr[t], false, b, (short)0, acc[t], false, false);
    }
    __syncthreads();
  }

  // epilogue 1: + b1, tanh, pack bf16 into smH
  // C/D layout: VGPR i -> M = i + khalf*8 (within tile), N = lane&15
  {
    float bias = b1[nt * 16 + mrow];
#pragma unroll
    for (int t = 0; t < 4; ++t) {
#pragma unroll
      for (int i = 0; i < 8; ++i) {
        float hv = tanhf(acc[t][i] + bias);
        int M = t * 16 + i + khalf * 8;
        smH[M * LD + nt * 16 + mrow] = (unsigned short)f2bf(hv);
      }
    }
  }
  __syncthreads();

  // ---- phase 2: out = h @ W2^T + b2 ----
  v8f acc2[4];
#pragma unroll
  for (int t = 0; t < 4; ++t) acc2[t] = (v8f)0.0f;

#pragma unroll
  for (int ks = 0; ks < 4; ++ks) {
    v16bf b = load_frag_ws(wsW2 + (ks * 8 + nt) * 256, lane);
    v16bf afr[4];
#pragma unroll
    for (int t = 0; t < 4; ++t)
      afr[t] = load_frag_lds(smH, (t * 16 + mrow) * LD + ks * 32 + khalf * 8);
#pragma unroll
    for (int t = 0; t < 4; ++t)
      acc2[t] = __builtin_amdgcn_wmma_f32_16x16x32_bf16(
          false, afr[t], false, b, (short)0, acc2[t], false, false);
  }

  {
    float bias = b2[nt * 16 + mrow];
#pragma unroll
    for (int t = 0; t < 4; ++t) {
#pragma unroll
      for (int i = 0; i < 8; ++i) {
        int gRow = rowbase + t * 16 + i + khalf * 8;
        if (gRow < Btot)
          __builtin_nontemporal_store(
              acc2[t][i] + bias,
              out + (long long)gRow * FEAT + nt * 16 + mrow);
      }
    }
  }
}

extern "C" void kernel_launch(void* const* d_in, const int* in_sizes, int n_in,
                              void* d_out, int out_size, void* d_ws, size_t ws_size,
                              hipStream_t stream) {
  (void)n_in; (void)out_size; (void)ws_size;
  const int*   nodes    = (const int*)d_in[0];
  const float* features = (const float*)d_in[1];
  const float* n0       = (const float*)d_in[2];
  const float* n1       = (const float*)d_in[3];
  const float* n2       = (const float*)d_in[4];
  const float* W1       = (const float*)d_in[5];
  const float* b1       = (const float*)d_in[6];
  const float* W2       = (const float*)d_in[7];
  const float* b2       = (const float*)d_in[8];
  float* out = (float*)d_out;
  int Btot = in_sizes[0];

  uint32_t* wsW1 = (uint32_t*)d_ws;           // 128 frags * 1 KB = 128 KB
  uint32_t* wsW2 = wsW1 + 128 * 256;          //  32 frags * 1 KB =  32 KB

  prep_w1<<<16, 256, 0, stream>>>(W1, wsW1);
  prep_w2<<<4, 256, 0, stream>>>(W2, wsW2);

  int grid = (Btot + MT - 1) / MT;
  encoder_main<<<grid, 256, 0, stream>>>(nodes, features, n0, n1, n2, b1, b2,
                                         wsW1, wsW2, out, Btot);
}
